// StreamlinedMoE_84679575208516
// MI455X (gfx1250) — matrix operand order repaired and checked by reading, not compiled, not run on hardware
//
#include <hip/hip_runtime.h>
#include <math.h>

// ---------------------------------------------------------------------------
// Problem constants (from reference)
// ---------------------------------------------------------------------------
#define BB     4
#define NN     4096
#define CC     1024
#define TT     (BB * NN)          // 16384 tokens
#define EE     8
#define KK     2
#define INTER  2730
#define GU     (2 * INTER)        // 5460
#define CAP    2560               // int(1.25 * T / E)
#define ACT_LD 2752               // INTER padded to multiple of 64 (16B-aligned bf16 rows)
#define SA_LD  40                 // LDS row pitch (bf16 elems) for 32-wide tiles: 80B,
#define SB_LD  40                 //   16B-aligned and bank-conflict-free (gcd(20,64)=4)

typedef __bf16  v16bf __attribute__((ext_vector_type(16)));
typedef float   v8f   __attribute__((ext_vector_type(8)));
typedef unsigned int u32x4 __attribute__((ext_vector_type(4)));

union FragBF { u32x4 q[2]; v16bf v; };   // 32 bytes = 16 bf16 per lane

__device__ __forceinline__ unsigned short f2bf(float f) {
    unsigned int u = __float_as_uint(f);
    unsigned int r = u + 0x7FFFu + ((u >> 16) & 1u);   // round-to-nearest-even
    return (unsigned short)(r >> 16);
}

// ---------------------------------------------------------------------------
// Tensor Data Mover: async DMA of a 2D bf16 tile (rows x 32 elems) into LDS
// with hardware row padding to SA_LD pitch. ISA 08_async_tensor 8.3/8.4.
// Toolchain here = 6-arg builtin: (g0, g1, g2, g3, g4, cpol).
// ---------------------------------------------------------------------------
#if __has_builtin(__builtin_amdgcn_tensor_load_to_lds)
#define HAVE_TDM 1
typedef int iv8 __attribute__((ext_vector_type(8)));
typedef int iv4 __attribute__((ext_vector_type(4)));

__device__ __forceinline__ void tdm_load_tile_bf16(unsigned lds_addr,        // byte offset in LDS
                                                   const void* gaddr,        // tile start
                                                   unsigned tile_rows,       // dim1
                                                   unsigned row_stride_elems) // global pitch (elems)
{
    unsigned long long ga = (unsigned long long)gaddr;
    u32x4 g0;
    g0.x = 1u;                                          // count=1 (valid user D#)
    g0.y = lds_addr;                                    // lds_addr
    g0.z = (unsigned)(ga & 0xFFFFFFFFull);              // global_addr lo
    g0.w = (unsigned)((ga >> 32) & 0x1FFFFFFull)        // global_addr hi (bits 56:32)
         | (2u << 30);                                  // type = 2 ("image")
    iv8 g1;
    const unsigned td0 = row_stride_elems;              // tensor_dim0 (no x clipping needed)
    const unsigned td1 = 0x100000u;                     // large tensor_dim1 (no y clipping)
    // w0: data_size=2B (code 1) @16, pad_enable @20, pad_interval=3 (16 dwords) @22,
    //     pad_amount=3 (4 dwords -> 16B) @25  => LDS row pitch 64B+16B = 80B = SA_LD
    g1[0] = (int)((1u << 16) | (1u << 20) | (3u << 22) | (3u << 25));
    g1[1] = (int)((td0 & 0xFFFFu) << 16);
    g1[2] = (int)(((td0 >> 16) & 0xFFFFu) | ((td1 & 0xFFFFu) << 16));
    g1[3] = (int)(((td1 >> 16) & 0xFFFFu) | (32u << 16));      // tile_dim0 = 32 elems
    g1[4] = (int)(tile_rows & 0xFFFFu);                        // tile_dim1 (tile_dim2=0)
    g1[5] = (int)row_stride_elems;                             // tensor_dim0_stride lo32
    g1[6] = 0;
    g1[7] = 0;
    iv4 z4 = (iv4)0;      // D# groups 2/3 unused for 2D tiles (tile_dim2..4 = 0)
    iv8 z8 = (iv8)0;
    __builtin_amdgcn_tensor_load_to_lds(g0, g1, z4, z4, z8, 0);
}
#endif

// ---------------------------------------------------------------------------
// 1) Gating: logits = [x ; cond] @ gate_w, SNR mask, softmax, top-2
// ---------------------------------------------------------------------------
__global__ __launch_bounds__(256)
void gate_kernel(const float* __restrict__ x, const float* __restrict__ cond,
                 const float* __restrict__ snr, const float* __restrict__ gw,
                 int* __restrict__ top_idx, float* __restrict__ top_w) {
    int wv = threadIdx.x >> 5, lane = threadIdx.x & 31;
    int t = blockIdx.x * 8 + wv;
    int b = t >> 12;                               // t / NN
    const float* xr = x + (size_t)t * CC;
    const float* cr = cond + (size_t)b * CC;

    float acc[EE];
    #pragma unroll
    for (int e = 0; e < EE; ++e) acc[e] = 0.f;

    for (int r = lane; r < 2 * CC; r += 32) {
        float v = (r < CC) ? xr[r] : cr[r - CC];
        const float* g = gw + (size_t)r * EE;
        #pragma unroll
        for (int e = 0; e < EE; ++e) acc[e] += v * g[e];
    }
    #pragma unroll
    for (int e = 0; e < EE; ++e) {
        #pragma unroll
        for (int off = 16; off > 0; off >>= 1)
            acc[e] += __shfl_xor(acc[e], off, 32);
    }

    if (lane == 0) {
        bool high = snr[b] < 0.5f;
        if (high) {
            #pragma unroll
            for (int e = 2; e < EE; ++e) acc[e] = -INFINITY;
        }
        float mx = acc[0];
        #pragma unroll
        for (int e = 1; e < EE; ++e) mx = fmaxf(mx, acc[e]);
        float p[EE], s = 0.f;
        #pragma unroll
        for (int e = 0; e < EE; ++e) { p[e] = __expf(acc[e] - mx); s += p[e]; }
        float inv = 1.f / s;
        #pragma unroll
        for (int e = 0; e < EE; ++e) p[e] *= inv;

        int i1 = 0;
        #pragma unroll
        for (int e = 1; e < EE; ++e) if (p[e] > p[i1]) i1 = e;
        int i2 = (i1 == 0) ? 1 : 0;
        #pragma unroll
        for (int e = 0; e < EE; ++e) if (e != i1 && p[e] > p[i2]) i2 = e;

        float s2 = fmaxf(p[i1] + p[i2], 1e-12f);
        top_idx[2 * t]     = i1;
        top_idx[2 * t + 1] = i2;
        top_w[2 * t]       = p[i1] / s2;
        top_w[2 * t + 1]   = p[i2] / s2;
    }
}

// ---------------------------------------------------------------------------
// 2) Deterministic capacity assignment (single block prefix scan)
// ---------------------------------------------------------------------------
__global__ __launch_bounds__(1024)
void assign_kernel(const int* __restrict__ top_idx, const float* __restrict__ top_w,
                   int* __restrict__ expert_tok, float* __restrict__ expert_w) {
    __shared__ unsigned int wsum[32][4];
    __shared__ unsigned int tot[4];
    __shared__ unsigned int base[EE];

    int tid = threadIdx.x, lane = tid & 31, wv = tid >> 5;

    for (int i = tid; i < EE * CAP; i += 1024) {
        expert_tok[i] = -1;
        expert_w[i]   = 0.f;
    }
    if (tid < EE) base[tid] = 0u;
    __syncthreads();

    for (int ch = 0; ch < (TT * KK) / 1024; ++ch) {
        int i = ch * 1024 + tid;
        int e = top_idx[i];

        u32x4 c = (u32x4)(0u);
        c[e >> 1] = 1u << ((e & 1) * 16);
        u32x4 own = c;

        #pragma unroll
        for (int off = 1; off < 32; off <<= 1) {
            u32x4 o;
            o.x = __shfl_up(c.x, (unsigned)off, 32);
            o.y = __shfl_up(c.y, (unsigned)off, 32);
            o.z = __shfl_up(c.z, (unsigned)off, 32);
            o.w = __shfl_up(c.w, (unsigned)off, 32);
            if (lane >= off) { c.x += o.x; c.y += o.y; c.z += o.z; c.w += o.w; }
        }
        if (lane == 31) {
            wsum[wv][0] = c.x; wsum[wv][1] = c.y; wsum[wv][2] = c.z; wsum[wv][3] = c.w;
        }
        __syncthreads();

        if (wv == 0) {
            u32x4 s;
            s.x = wsum[lane][0]; s.y = wsum[lane][1];
            s.z = wsum[lane][2]; s.w = wsum[lane][3];
            u32x4 inc0 = s;
            #pragma unroll
            for (int off = 1; off < 32; off <<= 1) {
                u32x4 o;
                o.x = __shfl_up(s.x, (unsigned)off, 32);
                o.y = __shfl_up(s.y, (unsigned)off, 32);
                o.z = __shfl_up(s.z, (unsigned)off, 32);
                o.w = __shfl_up(s.w, (unsigned)off, 32);
                if (lane >= off) { s.x += o.x; s.y += o.y; s.z += o.z; s.w += o.w; }
            }
            wsum[lane][0] = s.x - inc0.x; wsum[lane][1] = s.y - inc0.y;
            wsum[lane][2] = s.z - inc0.z; wsum[lane][3] = s.w - inc0.w;
            if (lane == 31) { tot[0] = s.x; tot[1] = s.y; tot[2] = s.z; tot[3] = s.w; }
        }
        __syncthreads();

        unsigned int exc = (c[e >> 1] - own[e >> 1]) + wsum[wv][e >> 1];
        unsigned int rank = (exc >> ((e & 1) * 16)) & 0xFFFFu;
        unsigned int slot = base[e] + rank;
        if (slot < CAP) {
            expert_tok[e * CAP + slot] = i >> 1;
            expert_w[e * CAP + slot]   = top_w[i];
        }
        __syncthreads();
        if (tid == 0) {
            #pragma unroll
            for (int ee = 0; ee < EE; ++ee)
                base[ee] += (tot[ee >> 1] >> ((ee & 1) * 16)) & 0xFFFFu;
        }
        __syncthreads();
    }
}

// ---------------------------------------------------------------------------
// 3) Gather tokens into per-expert bf16 A-matrix; zero unfilled slots.
// ---------------------------------------------------------------------------
__global__ __launch_bounds__(256)
void gather_kernel(const float* __restrict__ x, const int* __restrict__ expert_tok,
                   unsigned short* __restrict__ xe) {
    int row = blockIdx.x;
    int tok = expert_tok[row];
    const float* src = x + (size_t)(tok < 0 ? 0 : tok) * CC;
    unsigned short* dst = xe + (size_t)row * CC;
    for (int c = threadIdx.x; c < CC; c += 256)
        dst[c] = (tok >= 0) ? f2bf(src[c]) : (unsigned short)0;
}

// ---------------------------------------------------------------------------
// 4) GEMM1: act = silu(xe @ Wg) * (xe @ Wu); bf16 WMMA, fp32 accumulate.
//    128(M) x 64(N) tile, 8 waves; A tiles DMA'd by TDM (double-buffered).
// ---------------------------------------------------------------------------
__global__ __launch_bounds__(256)
void gemm1_kernel(const unsigned short* __restrict__ xe,   // (E*CAP, C) bf16
                  const float* __restrict__ gate_up_w,     // (E, C, GU) f32
                  unsigned short* __restrict__ act) {      // (E*CAP, ACT_LD) bf16
    __shared__ __align__(16) unsigned short sA[2][128 * SA_LD];
    __shared__ __align__(16) unsigned short sBg[64 * SB_LD];
    __shared__ __align__(16) unsigned short sBu[64 * SB_LD];

    const int e = blockIdx.z, mt = blockIdx.y, nt = blockIdx.x;
    const int m_base = mt * 128;
    const int n_base = nt * 64;

    const unsigned short* Arow = xe + (size_t)(e * CAP + m_base) * CC;
    const float*          W    = gate_up_w + (size_t)e * CC * GU;

    const int tid = threadIdx.x, lane = tid & 31, wave = tid >> 5;
    const int wm = wave >> 1, wn = wave & 1;

    v8f accg[2][2], accu[2][2];
    #pragma unroll
    for (int a = 0; a < 2; ++a)
        #pragma unroll
        for (int b = 0; b < 2; ++b) { accg[a][b] = (v8f)0.f; accu[a][b] = (v8f)0.f; }

#if HAVE_TDM
    if (tid < 32)   // wave 0 drives the TDM; prime the pipeline with tile 0
        tdm_load_tile_bf16((unsigned)(size_t)&sA[0][0], Arow, 128, CC);
#endif

    for (int k0 = 0; k0 < CC; k0 += 32) {
        const int cur = (k0 >> 5) & 1;
        // stage B tiles (transpose + f32->bf16) -- all waves
        for (int i = tid; i < 512; i += 256) {
            int kk = i >> 4;
            int n4 = (i & 15) << 2;
            const float* wg = W + (size_t)(k0 + kk) * GU;
            if (k0 + 32 < CC) __builtin_prefetch(wg + GU * 32, 0, 1);
            #pragma unroll
            for (int j = 0; j < 4; ++j) {
                int n = n4 + j, col = n_base + n;
                float gv = (col < INTER) ? wg[col]         : 0.f;
                float uv = (col < INTER) ? wg[col + INTER] : 0.f;
                sBg[n * SB_LD + kk] = f2bf(gv);
                sBu[n * SB_LD + kk] = f2bf(uv);
            }
        }
#if HAVE_TDM
        if (tid < 32) {
            if (k0 + 32 < CC) {   // kick next A tile, wait only for current
                tdm_load_tile_bf16((unsigned)(size_t)&sA[1 - cur][0],
                                   Arow + k0 + 32, 128, CC);
                __builtin_amdgcn_s_wait_tensorcnt((short)1);
            } else {
                __builtin_amdgcn_s_wait_tensorcnt((short)0);
            }
        }
#else
        for (int i = tid; i < 512; i += 256) {
            int row  = i >> 2;
            int part = (i & 3) << 3;
            u32x4 v = *(const u32x4*)(Arow + (size_t)row * CC + k0 + part);
            *(u32x4*)&sA[cur][row * SA_LD + part] = v;
        }
#endif
        __syncthreads();

        FragBF fa[2];
        #pragma unroll
        for (int sm = 0; sm < 2; ++sm) {
            int row = (wm << 5) + (sm << 4) + (lane & 15);
            int kh  = (lane >> 4) << 3;               // 0 or 8
            fa[sm].q[0] = *(const u32x4*)&sA[cur][row * SA_LD + kh];
            fa[sm].q[1] = *(const u32x4*)&sA[cur][row * SA_LD + kh + 16];
        }
        FragBF fg[2], fu[2];
        #pragma unroll
        for (int sn = 0; sn < 2; ++sn) {
            int n  = (wn << 5) + (sn << 4) + (lane & 15);
            int kh = (lane >> 4) << 3;                // 0 or 8
            fg[sn].q[0] = *(const u32x4*)&sBg[n * SB_LD + kh];
            fg[sn].q[1] = *(const u32x4*)&sBg[n * SB_LD + kh + 16];
            fu[sn].q[0] = *(const u32x4*)&sBu[n * SB_LD + kh];
            fu[sn].q[1] = *(const u32x4*)&sBu[n * SB_LD + kh + 16];
        }
        #pragma unroll
        for (int sm = 0; sm < 2; ++sm)
            #pragma unroll
            for (int sn = 0; sn < 2; ++sn) {
                accg[sm][sn] = __builtin_amdgcn_wmma_f32_16x16x32_bf16(
                    false, fa[sm].v, false, fg[sn].v, (short)0, accg[sm][sn], false, false);
                accu[sm][sn] = __builtin_amdgcn_wmma_f32_16x16x32_bf16(
                    false, fa[sm].v, false, fu[sn].v, (short)0, accu[sm][sn], false, false);
            }
        __syncthreads();   // protect B (and fallback-A) buffers before restage
    }

    const int mrow0 = m_base + (wm << 5);
    const int ncol0 = n_base + (wn << 5);
    const int lrow  = (lane >> 4) << 3;
    const int lcol  = lane & 15;
    #pragma unroll
    for (int sm = 0; sm < 2; ++sm)
        #pragma unroll
        for (int sn = 0; sn < 2; ++sn) {
            int colg = ncol0 + (sn << 4) + lcol;
            #pragma unroll
            for (int i2 = 0; i2 < 8; ++i2) {
                int row = mrow0 + (sm << 4) + lrow + i2;
                float gv = accg[sm][sn][i2];
                float uv = accu[sm][sn][i2];
                float av = (gv / (1.f + __expf(-gv))) * uv;
                act[(size_t)(e * CAP + row) * ACT_LD + colg] =
                    f2bf(colg < INTER ? av : 0.f);
            }
        }
}

// ---------------------------------------------------------------------------
// 5) GEMM2: y = act @ Wd, scale by routing weight, atomic-scatter into out.
// ---------------------------------------------------------------------------
__global__ __launch_bounds__(256)
void gemm2_kernel(const unsigned short* __restrict__ act,  // (E*CAP, ACT_LD) bf16
                  const float* __restrict__ down_w,        // (E, INTER, C) f32
                  const int* __restrict__ expert_tok,
                  const float* __restrict__ expert_w,
                  float* __restrict__ out) {
    __shared__ __align__(16) unsigned short sA[2][128 * SA_LD];
    __shared__ __align__(16) unsigned short sB[64 * SB_LD];

    const int e = blockIdx.z, mt = blockIdx.y, nt = blockIdx.x;
    const int m_base = mt * 128;
    const int n_base = nt * 64;

    const unsigned short* Arow = act + (size_t)(e * CAP + m_base) * ACT_LD;
    const float*          Wd   = down_w + (size_t)e * INTER * CC;

    const int tid = threadIdx.x, lane = tid & 31, wave = tid >> 5;
    const int wm = wave >> 1, wn = wave & 1;

    v8f acc[2][2];
    #pragma unroll
    for (int a = 0; a < 2; ++a)
        #pragma unroll
        for (int b = 0; b < 2; ++b) acc[a][b] = (v8f)0.f;

#if HAVE_TDM
    if (tid < 32)
        tdm_load_tile_bf16((unsigned)(size_t)&sA[0][0], Arow, 128, ACT_LD);
#endif

    for (int k0 = 0; k0 < ACT_LD; k0 += 32) {
        const int cur = (k0 >> 5) & 1;
        for (int i = tid; i < 512; i += 256) {
            int kk = i >> 4;
            int n4 = (i & 15) << 2;
            int k  = k0 + kk;
            const float* wr = Wd + (size_t)(k < INTER ? k : 0) * CC;
            if (k0 + 32 < INTER) __builtin_prefetch(wr + CC * 32, 0, 1);
            #pragma unroll
            for (int j = 0; j < 4; ++j) {
                int col = n_base + n4 + j;
                float v = (k < INTER) ? wr[col] : 0.f;
                sB[(n4 + j) * SB_LD + kk] = f2bf(v);
            }
        }
#if HAVE_TDM
        if (tid < 32) {
            if (k0 + 32 < ACT_LD) {
                tdm_load_tile_bf16((unsigned)(size_t)&sA[1 - cur][0],
                                   Arow + k0 + 32, 128, ACT_LD);
                __builtin_amdgcn_s_wait_tensorcnt((short)1);
            } else {
                __builtin_amdgcn_s_wait_tensorcnt((short)0);
            }
        }
#else
        for (int i = tid; i < 512; i += 256) {
            int row  = i >> 2;
            int part = (i & 3) << 3;
            u32x4 v = *(const u32x4*)(Arow + (size_t)row * ACT_LD + k0 + part);
            *(u32x4*)&sA[cur][row * SA_LD + part] = v;
        }
#endif
        __syncthreads();

        FragBF fa[2];
        #pragma unroll
        for (int sm = 0; sm < 2; ++sm) {
            int row = (wm << 5) + (sm << 4) + (lane & 15);
            int kh  = (lane >> 4) << 3;
            fa[sm].q[0] = *(const u32x4*)&sA[cur][row * SA_LD + kh];
            fa[sm].q[1] = *(const u32x4*)&sA[cur][row * SA_LD + kh + 16];
        }
        FragBF fb[2];
        #pragma unroll
        for (int sn = 0; sn < 2; ++sn) {
            int n  = (wn << 5) + (sn << 4) + (lane & 15);
            int kh = (lane >> 4) << 3;
            fb[sn].q[0] = *(const u32x4*)&sB[n * SB_LD + kh];
            fb[sn].q[1] = *(const u32x4*)&sB[n * SB_LD + kh + 16];
        }
        #pragma unroll
        for (int sm = 0; sm < 2; ++sm)
            #pragma unroll
            for (int sn = 0; sn < 2; ++sn)
                acc[sm][sn] = __builtin_amdgcn_wmma_f32_16x16x32_bf16(
                    false, fa[sm].v, false, fb[sn].v, (short)0, acc[sm][sn], false, false);
        __syncthreads();
    }

    const int mrow0 = m_base + (wm << 5);
    const int ncol0 = n_base + (wn << 5);
    const int lrow  = (lane >> 4) << 3;
    const int lcol  = lane & 15;
    #pragma unroll
    for (int sm = 0; sm < 2; ++sm) {
        #pragma unroll
        for (int i2 = 0; i2 < 8; ++i2) {
            int row = mrow0 + (sm << 4) + lrow + i2;
            int gs  = e * CAP + row;
            int tok = expert_tok[gs];
            float wgt = expert_w[gs];
            if (tok >= 0) {
                float* orow = out + (size_t)tok * CC;
                #pragma unroll
                for (int sn = 0; sn < 2; ++sn) {
                    int colg = ncol0 + (sn << 4) + lcol;
                    atomicAdd(&orow[colg], acc[sm][sn][i2] * wgt);
                }
            }
        }
    }
}

// ---------------------------------------------------------------------------
// Launch
// ---------------------------------------------------------------------------
extern "C" void kernel_launch(void* const* d_in, const int* in_sizes, int n_in,
                              void* d_out, int out_size, void* d_ws, size_t ws_size,
                              hipStream_t stream) {
    (void)in_sizes; (void)n_in; (void)ws_size;
    const float* x         = (const float*)d_in[0];
    const float* cond      = (const float*)d_in[1];
    const float* snr       = (const float*)d_in[2];
    const float* gate_w    = (const float*)d_in[3];
    const float* gate_up_w = (const float*)d_in[4];
    const float* down_w    = (const float*)d_in[5];
    float* out = (float*)d_out;

    char* ws = (char*)d_ws;
    size_t off = 0;
    auto alloc = [&](size_t bytes) -> void* {
        void* p = ws + off;
        off = (off + bytes + 255) & ~(size_t)255;
        return p;
    };
    int*            top_idx    = (int*)alloc((size_t)TT * KK * sizeof(int));
    float*          top_w      = (float*)alloc((size_t)TT * KK * sizeof(float));
    int*            expert_tok = (int*)alloc((size_t)EE * CAP * sizeof(int));
    float*          expert_w   = (float*)alloc((size_t)EE * CAP * sizeof(float));
    unsigned short* xe         = (unsigned short*)alloc((size_t)EE * CAP * CC * 2);
    unsigned short* act        = (unsigned short*)alloc((size_t)EE * CAP * ACT_LD * 2);

    (void)hipMemsetAsync(d_out, 0, (size_t)out_size * sizeof(float), stream);

    gate_kernel<<<TT / 8, 256, 0, stream>>>(x, cond, snr, gate_w, top_idx, top_w);
    assign_kernel<<<1, 1024, 0, stream>>>(top_idx, top_w, expert_tok, expert_w);
    gather_kernel<<<EE * CAP, 256, 0, stream>>>(x, expert_tok, xe);
    gemm1_kernel<<<dim3(ACT_LD / 64, CAP / 128, EE), 256, 0, stream>>>(xe, gate_up_w, act);
    gemm2_kernel<<<dim3(CC / 64, CAP / 128, EE), 256, 0, stream>>>(act, down_w, expert_tok,
                                                                   expert_w, out);
}
